// GAT_71296457113908
// MI455X (gfx1250) — matrix operand order repaired and checked by reading, not compiled, not run on hardware
//
#include <hip/hip_runtime.h>
#include <hip/hip_bf16.h>

typedef __attribute__((ext_vector_type(16))) _Float16 v16h;
typedef __attribute__((ext_vector_type(2)))  _Float16 v2h;
typedef __attribute__((ext_vector_type(8)))  float    v8f;

#define HEADS 2
#define OUTD  64
#define CH    128          // HEADS*OUTD
#define NEG_SLOPE 0.2f

// ---------------------------------------------------------------------------
// utility kernels
// ---------------------------------------------------------------------------
__global__ void k_fill_f32(float* __restrict__ p, float v, long long n) {
    long long i = (long long)blockIdx.x * blockDim.x + threadIdx.x;
    if (i < n) p[i] = v;
}

__global__ void k_fill_f16(_Float16* __restrict__ p, long long n) {
    long long i = (long long)blockIdx.x * blockDim.x + threadIdx.x;
    if (i < n) p[i] = (_Float16)0.0f;
}

__global__ void k_f32_to_f16(const float* __restrict__ x, _Float16* __restrict__ y,
                             long long n) {
    long long i = (long long)blockIdx.x * blockDim.x + threadIdx.x;
    if (i < n) y[i] = (_Float16)x[i];
}

// ---------------------------------------------------------------------------
// GEMM: D[M,128] = A[M,128] * B[128,128]; f16 inputs, f32 accumulate.
// One wave per 16x16 output tile, 8 waves (N-tiles) per block, K-loop of 4
// v_wmma_f32_16x16x32_f16.
//   A frag (16x32 f16): lane&15 = M row; lane>=16 selects K-subgroup +8;
//                       VGPR pairs pack K per the CDNA5 ISA layout.
//   B frag (32x16 f16): lane&15 = N col; lane>=16 selects K-subgroup +8; elements hold K
//                       in A's order -> 16 strided f16 loads per lane.
//   C/D   (16x16 f32):  lane&15 = N col; VGPR v = row v (+8 for lanes>=16).
// ---------------------------------------------------------------------------
__global__ __launch_bounds__(256)
void k_gemm128(const _Float16* __restrict__ A, const _Float16* __restrict__ B,
               float* __restrict__ D) {
    const int mtile = blockIdx.x;
    const int ntile = threadIdx.x >> 5;         // 0..7
    const int lane  = threadIdx.x & 31;
    const int mrow  = (mtile << 4) + (lane & 15);
    const int kbA   = (lane >> 4) << 3;                     // 0 or 8
    const int kB    = (lane & 15) | ((lane >> 4) << 4);     // 0..31

    const _Float16* arow = A + (size_t)mrow * CH;
    v8f acc = {};
#pragma unroll
    for (int kc = 0; kc < CH; kc += 32) {
        v16h af;
#pragma unroll
        for (int v = 0; v < 8; ++v) {
            const int k = kc + ((v >> 2) << 4) + kbA + ((v & 3) << 1);
            v2h p = *(const v2h*)(arow + k);
            af[2 * v]     = p[0];
            af[2 * v + 1] = p[1];
        }
        v16h bf;
#pragma unroll
        for (int s = 0; s < 16; ++s) {
            const int k = kc + ((s >> 3) << 4) + kbA + (s & 7);
            bf[s] = B[(size_t)k * CH + (ntile << 4) + (lane & 15)];
        }
        acc = __builtin_amdgcn_wmma_f32_16x16x32_f16(false, af, false, bf,
                                                     (short)0, acc, false, false);
    }
    const int col   = (ntile << 4) + (lane & 15);
    const int rbase = (mtile << 4) + ((lane >> 4) << 3);
#pragma unroll
    for (int v = 0; v < 8; ++v)
        D[(size_t)(rbase + v) * CH + col] = acc[v];
}

// ---------------------------------------------------------------------------
// per-node attention logits: a_s[n,h] = sum_d h[n,h,d]*as[h,d]  (same for a_d)
// one block (128 threads) per node; 64-wide LDS tree reduction per head.
// ---------------------------------------------------------------------------
__global__ __launch_bounds__(128)
void k_alpha(const float* __restrict__ h, const float* __restrict__ as,
             const float* __restrict__ ad, float* __restrict__ a_s,
             float* __restrict__ a_d) {
    const int n = blockIdx.x, t = threadIdx.x;
    const int head = t >> 6, d = t & 63;
    const float hv = h[(size_t)n * CH + t];
    __shared__ float ss[CH], sd[CH];
    ss[t] = hv * as[head * OUTD + d];
    sd[t] = hv * ad[head * OUTD + d];
    __syncthreads();
    for (int off = 32; off > 0; off >>= 1) {
        if (d < off) { ss[t] += ss[t + off]; sd[t] += sd[t + off]; }
        __syncthreads();
    }
    if (d == 0) {
        a_s[n * HEADS + head] = ss[t];
        a_d[n * HEADS + head] = sd[t];
    }
}

// ---------------------------------------------------------------------------
// edge helpers
// ---------------------------------------------------------------------------
__device__ __forceinline__ void edge_sd(const int* __restrict__ ei, int E, int e,
                                        int& s, int& d) {
    if (e < E) { s = ei[e]; d = ei[E + e]; }
    else       { s = d = e - E; }               // appended self loops
}

__device__ __forceinline__ float lrelu(float a) {
    return a > 0.0f ? a : NEG_SLOPE * a;
}

__device__ __forceinline__ void atomicMaxF32(float* addr, float val) {
    unsigned int* ua  = (unsigned int*)addr;
    unsigned int  old = *ua;
    while (__uint_as_float(old) < val) {
        unsigned int assumed = old;
        old = atomicCAS(ua, assumed, __float_as_uint(val));
        if (old == assumed) break;
    }
}

// pass 1: segment max of leaky-relu logits over dst
__global__ void k_edge_max(const int* __restrict__ ei, int E, int EN,
                           const float* __restrict__ a_s,
                           const float* __restrict__ a_d,
                           float* __restrict__ amax) {
    const long long i = (long long)blockIdx.x * blockDim.x + threadIdx.x;
    if (i >= (long long)EN * HEADS) return;
    const int e = (int)(i >> 1), h = (int)(i & 1);
    int s, d; edge_sd(ei, E, e, s, d);
    atomicMaxF32(&amax[d * HEADS + h], lrelu(a_s[s * HEADS + h] + a_d[d * HEADS + h]));
}

// pass 2: segment sum of exp(alpha - amax)
__global__ void k_edge_sum(const int* __restrict__ ei, int E, int EN,
                           const float* __restrict__ a_s,
                           const float* __restrict__ a_d,
                           const float* __restrict__ amax,
                           float* __restrict__ denom) {
    const long long i = (long long)blockIdx.x * blockDim.x + threadIdx.x;
    if (i >= (long long)EN * HEADS) return;
    const int e = (int)(i >> 1), h = (int)(i & 1);
    int s, d; edge_sd(ei, E, e, s, d);
    const float a = lrelu(a_s[s * HEADS + h] + a_d[d * HEADS + h]);
    atomicAdd(&denom[d * HEADS + h], __expf(a - amax[d * HEADS + h]));
}

// pass 3: out[dst] += nq * attn * h[src]   (128 threads per edge, 1 feat each)
__global__ __launch_bounds__(256)
void k_edge_msg(const int* __restrict__ ei, int E, int EN,
                const float* __restrict__ ecent, const float* __restrict__ ncent,
                const float* __restrict__ a_s, const float* __restrict__ a_d,
                const float* __restrict__ amax, const float* __restrict__ denom,
                const float* __restrict__ h, float* __restrict__ out) {
    const long long i = (long long)blockIdx.x * blockDim.x + threadIdx.x;
    if (i >= (long long)EN * CH) return;
    const int e  = (int)(i >> 7);
    const int t  = (int)(i & (CH - 1));
    const int hd = t >> 6;
    int s, d; edge_sd(ei, E, e, s, d);
    const float a    = lrelu(a_s[s * HEADS + hd] + a_d[d * HEADS + hd]);
    const float ex   = __expf(a - amax[d * HEADS + hd]);
    const float attn = ex / (denom[d * HEADS + hd] + 1e-16f);
    const float nq   = (e < E) ? ecent[e] : ncent[e - E];
    atomicAdd(&out[(size_t)d * CH + t], nq * attn * h[(size_t)s * CH + t]);
}

__global__ void k_bias_relu(float* __restrict__ o, const float* __restrict__ b,
                            long long n) {
    long long i = (long long)blockIdx.x * blockDim.x + threadIdx.x;
    if (i >= n) return;
    const float v = o[i] + b[i & (CH - 1)];
    o[i] = v > 0.0f ? v : 0.0f;
}

// ---------------------------------------------------------------------------
// global mean pool + classifier
// ---------------------------------------------------------------------------
__global__ void k_pool_acc(const float* __restrict__ act,
                           const int* __restrict__ batch,
                           float* __restrict__ pooled, long long n) {
    long long i = (long long)blockIdx.x * blockDim.x + threadIdx.x;
    if (i >= n) return;
    const int node = (int)(i >> 7), c = (int)(i & (CH - 1));
    atomicAdd(&pooled[(size_t)batch[node] * CH + c], act[i]);
}

__global__ void k_cnt(const int* __restrict__ batch, float* __restrict__ cnt, int n) {
    int i = blockIdx.x * blockDim.x + threadIdx.x;
    if (i < n) atomicAdd(&cnt[batch[i]], 1.0f);
}

__global__ void k_classify(const float* __restrict__ pooled,
                           const float* __restrict__ cnt,
                           const float* __restrict__ Wc,
                           const float* __restrict__ bc,
                           float* __restrict__ out, int G) {
    int i = blockIdx.x * blockDim.x + threadIdx.x;
    if (i >= G * 10) return;
    const int g = i / 10, j = i % 10;
    const float inv = 1.0f / fmaxf(cnt[g], 1.0f);
    float s = 0.0f;
    for (int k = 0; k < CH; ++k) s += pooled[(size_t)g * CH + k] * Wc[k * 10 + j];
    out[i] = s * inv + bc[j];
}

// ---------------------------------------------------------------------------
// launch
// ---------------------------------------------------------------------------
static inline unsigned nblk(long long n, int t) { return (unsigned)((n + t - 1) / t); }
static inline size_t   aln(size_t o)            { return (o + 255) & ~(size_t)255; }

extern "C" void kernel_launch(void* const* d_in, const int* in_sizes, int n_in,
                              void* d_out, int out_size, void* d_ws, size_t ws_size,
                              hipStream_t stream) {
    const float* x     = (const float*)d_in[0];
    const int*   ei    = (const int*)  d_in[1];
    const int*   batch = (const int*)  d_in[2];
    const float* ncent = (const float*)d_in[3];
    const float* ecent = (const float*)d_in[4];
    const float* W[2]  = { (const float*)d_in[5],  (const float*)d_in[9]  };
    const float* AS[2] = { (const float*)d_in[6],  (const float*)d_in[10] };
    const float* AD[2] = { (const float*)d_in[7],  (const float*)d_in[11] };
    const float* BI[2] = { (const float*)d_in[8],  (const float*)d_in[12] };
    const float* Wc    = (const float*)d_in[13];
    const float* bc    = (const float*)d_in[14];

    const int N    = in_sizes[3];
    const int E    = in_sizes[4];
    const int EN   = E + N;
    const int G    = out_size / 10;
    const int Npad = (N + 15) & ~15;

    // workspace layout (each buffer 256B-aligned)
    char* w = (char*)d_ws;
    size_t off = 0;
    float*    h_lin  = (float*)(w + off);    off = aln(off + (size_t)Npad * CH * 4);
    float*    oacc   = (float*)(w + off);    off = aln(off + (size_t)N    * CH * 4);
    _Float16* x16    = (_Float16*)(w + off); off = aln(off + (size_t)Npad * CH * 2);
    _Float16* w16    = (_Float16*)(w + off); off = aln(off + (size_t)CH * CH * 2);
    float*    a_s    = (float*)(w + off);    off = aln(off + (size_t)N * HEADS * 4);
    float*    a_d    = (float*)(w + off);    off = aln(off + (size_t)N * HEADS * 4);
    float*    amax   = (float*)(w + off);    off = aln(off + (size_t)N * HEADS * 4);
    float*    denom  = (float*)(w + off);    off = aln(off + (size_t)N * HEADS * 4);
    float*    pooled = (float*)(w + off);    off = aln(off + (size_t)G * CH * 4);
    float*    cnt    = (float*)(w + off);    off = aln(off + (size_t)G * 4);
    (void)ws_size; (void)n_in;

    const long long nNC = (long long)N * CH;
    const long long nEH = (long long)EN * HEADS;
    const long long nEC = (long long)EN * CH;

    for (int layer = 0; layer < 2; ++layer) {
        const float* xin = layer ? oacc : x;
        // stage f16 operands for WMMA
        k_f32_to_f16<<<nblk(nNC, 256), 256, 0, stream>>>(xin, x16, nNC);
        if (Npad > N)
            k_fill_f16<<<nblk((long long)(Npad - N) * CH, 256), 256, 0, stream>>>(
                x16 + (size_t)N * CH, (long long)(Npad - N) * CH);
        k_f32_to_f16<<<nblk(CH * CH, 256), 256, 0, stream>>>(W[layer], w16, CH * CH);
        // h = x @ W   (v_wmma_f32_16x16x32_f16)
        k_gemm128<<<Npad / 16, 256, 0, stream>>>(x16, w16, h_lin);
        // per-node logits
        k_alpha<<<N, 128, 0, stream>>>(h_lin, AS[layer], AD[layer], a_s, a_d);
        // segment softmax (3 edge passes) + message scatter
        k_fill_f32<<<nblk((long long)N * HEADS, 256), 256, 0, stream>>>(amax, -1e30f, (long long)N * HEADS);
        k_fill_f32<<<nblk((long long)N * HEADS, 256), 256, 0, stream>>>(denom, 0.0f, (long long)N * HEADS);
        k_edge_max<<<nblk(nEH, 256), 256, 0, stream>>>(ei, E, EN, a_s, a_d, amax);
        k_edge_sum<<<nblk(nEH, 256), 256, 0, stream>>>(ei, E, EN, a_s, a_d, amax, denom);
        k_fill_f32<<<nblk(nNC, 256), 256, 0, stream>>>(oacc, 0.0f, nNC);
        k_edge_msg<<<nblk(nEC, 256), 256, 0, stream>>>(ei, E, EN, ecent, ncent,
                                                       a_s, a_d, amax, denom,
                                                       h_lin, oacc);
        k_bias_relu<<<nblk(nNC, 256), 256, 0, stream>>>(oacc, BI[layer], nNC);
    }

    // mean pool per graph + classifier
    k_fill_f32<<<nblk((long long)G * CH, 256), 256, 0, stream>>>(pooled, 0.0f, (long long)G * CH);
    k_fill_f32<<<nblk(G, 256), 256, 0, stream>>>(cnt, 0.0f, G);
    k_pool_acc<<<nblk(nNC, 256), 256, 0, stream>>>(oacc, batch, pooled, nNC);
    k_cnt<<<nblk(N, 256), 256, 0, stream>>>(batch, cnt, N);
    k_classify<<<nblk((long long)G * 10, 256), 256, 0, stream>>>(pooled, cnt, Wc, bc,
                                                                 (float*)d_out, G);
}
